// Attention_19576460935678
// MI455X (gfx1250) — compile-verified, matrix operand order checked by reading, not run on hardware
//
#include <hip/hip_runtime.h>
#include <hip/hip_bf16.h>

// ---------------- problem constants ----------------
#define BB   4
#define NN   2048
#define CC   768
#define HH   12
#define DD   64
#define MTOT (BB * NN)        // 8192 rows
#define QKVN (3 * CC)         // 2304

typedef __bf16 bf16_t;
typedef __bf16 v16bf __attribute__((ext_vector_type(16)));
typedef __bf16 v8bf  __attribute__((ext_vector_type(8)));
typedef float  v8f   __attribute__((ext_vector_type(8)));
typedef unsigned int u32;

__device__ __forceinline__ v8f wmma_bf16(v16bf a, v16bf b, v8f c) {
    // (neg_a, A, neg_b, B, c_mod, C, reuse_a, reuse_b)
    return __builtin_amdgcn_wmma_f32_16x16x32_bf16(false, a, false, b, (short)0, c, false, false);
}

// Operand fragment for 16x16x32 bf16 WMMA from a tile where one row/col's 32
// K-values are CONTIGUOUS. Lane L covers row (L&15); element i is
// K = (i>>3)*16 + (L>>4)*8 + (i&7)  ->  two 16B contiguous ds_load_b128.
__device__ __forceinline__ v16bf load_frag(const bf16_t* rowptr, int lh) {
    v8bf lo = *(const v8bf*)(rowptr + lh * 8);
    v8bf hi = *(const v8bf*)(rowptr + 16 + lh * 8);
    v16bf f;
#pragma unroll
    for (int i = 0; i < 8; ++i) { f[i] = lo[i]; f[i + 8] = hi[i]; }
    return f;
}

__device__ __forceinline__ u32 pack_bf16(float a, float b) {
    bf16_t h0 = (bf16_t)a, h1 = (bf16_t)b;
    return ((u32) * (unsigned short*)&h0) | (((u32) * (unsigned short*)&h1) << 16);
}

// ---------------- fp32 -> bf16 conversion (4 elems/thread) ----------------
__global__ __launch_bounds__(256) void cvt_bf16_kernel(const float* __restrict__ src,
                                                       bf16_t* __restrict__ dst, int n) {
    int i = (blockIdx.x * 256 + threadIdx.x) * 4;
    if (i < n) {
        float4 f = *(const float4*)(src + i);
        ((u32*)(dst + i))[0] = pack_bf16(f.x, f.y);
        ((u32*)(dst + i))[1] = pack_bf16(f.z, f.w);
    }
}

// ---------------- fp32 [768][N] -> bf16 transposed [N][768] ----------------
// Coalesced f32 reads across n; 8B writes strided by 768*2 (one-off pass).
__global__ __launch_bounds__(256) void cvt_t_kernel(const float* __restrict__ src,
                                                    bf16_t* __restrict__ dst, int N) {
    int n = blockIdx.x * 256 + threadIdx.x;
    int k = blockIdx.y * 4;
    float f0 = src[(size_t)(k + 0) * N + n];
    float f1 = src[(size_t)(k + 1) * N + n];
    float f2 = src[(size_t)(k + 2) * N + n];
    float f3 = src[(size_t)(k + 3) * N + n];
    u32* d = (u32*)(dst + (size_t)n * CC + k);
    d[0] = pack_bf16(f0, f1);
    d[1] = pack_bf16(f2, f3);
}

// ---------------- tiled bf16 GEMM:  [MTOT x 768] @ [768 x NDIM] ----------------
// BM=256, BN=64, BK=64. 8 waves; wave w owns rows [w*32, w*32+32), all 64 cols.
// B arrives PRE-TRANSPOSED [NDIM][768], so every tile fill is a b128 copy.
// MODE 0: scatter into q/k [B][H][N][D] (q pre-scaled by 8) and v [B][H][D][N].
// MODE 1: write f32 to outf.
template <int NDIM, int MODE>
__global__ __launch_bounds__(256) void gemm_bf16_kernel(const bf16_t* __restrict__ A,
                                                        const bf16_t* __restrict__ BwT,
                                                        bf16_t* __restrict__ q,
                                                        bf16_t* __restrict__ k,
                                                        bf16_t* __restrict__ v,
                                                        float* __restrict__ outf) {
    __shared__ __align__(16) bf16_t As[256 * 64];   // [m][k]  32 KB
    __shared__ __align__(16) bf16_t Bt[64 * 64];    // [n][k]   8 KB

    const int tid  = threadIdx.x;
    const int lane = tid & 31;
    const int wid  = tid >> 5;      // 0..7: row group of 32
    const int ln   = lane & 15;
    const int lh   = lane >> 4;
    const int m0   = blockIdx.y * 256;
    const int n0   = blockIdx.x * 64;

    v8f acc[2][4];
#pragma unroll
    for (int i = 0; i < 2; ++i)
#pragma unroll
        for (int j = 0; j < 4; ++j)
            acc[i][j] = (v8f){0.f, 0.f, 0.f, 0.f, 0.f, 0.f, 0.f, 0.f};

    const uint4* Ag4 = (const uint4*)A;     // row stride CC/8 uint4
    const uint4* Bg4 = (const uint4*)BwT;   // row stride CC/8 uint4

    for (int k0 = 0; k0 < CC; k0 += 64) {
        // A tile: 256x64 bf16 = 2048 uint4
#pragma unroll
        for (int i = 0; i < 8; ++i) {
            int t = tid + i * 256;                 // t = r*8 + cu
            int r = t >> 3, cu = t & 7;
            ((uint4*)As)[t] = Ag4[(size_t)(m0 + r) * (CC / 8) + (k0 >> 3) + cu];
        }
        // B tile (already transposed in global): 64x64 bf16 = 512 uint4
#pragma unroll
        for (int i = 0; i < 2; ++i) {
            int t = tid + i * 256;
            int r = t >> 3, cu = t & 7;
            ((uint4*)Bt)[t] = Bg4[(size_t)(n0 + r) * (CC / 8) + (k0 >> 3) + cu];
        }
        __syncthreads();

#pragma unroll
        for (int ks = 0; ks < 2; ++ks) {
            int kb = ks * 32;
            v16bf a0 = load_frag(As + (wid * 32 + ln) * 64 + kb, lh);
            v16bf a1 = load_frag(As + (wid * 32 + 16 + ln) * 64 + kb, lh);
            v16bf b0 = load_frag(Bt + (ln) * 64 + kb, lh);
            v16bf b1 = load_frag(Bt + (16 + ln) * 64 + kb, lh);
            v16bf b2 = load_frag(Bt + (32 + ln) * 64 + kb, lh);
            v16bf b3 = load_frag(Bt + (48 + ln) * 64 + kb, lh);
            acc[0][0] = wmma_bf16(a0, b0, acc[0][0]);
            acc[0][1] = wmma_bf16(a0, b1, acc[0][1]);
            acc[0][2] = wmma_bf16(a0, b2, acc[0][2]);
            acc[0][3] = wmma_bf16(a0, b3, acc[0][3]);
            acc[1][0] = wmma_bf16(a1, b0, acc[1][0]);
            acc[1][1] = wmma_bf16(a1, b1, acc[1][1]);
            acc[1][2] = wmma_bf16(a1, b2, acc[1][2]);
            acc[1][3] = wmma_bf16(a1, b3, acc[1][3]);
        }
        __syncthreads();
    }

    // epilogue: C/D layout — lane gives N = (lane&15), M = r + (lane>>4)*8
    const int bb = m0 >> 11;              // block never crosses a batch boundary
#pragma unroll
    for (int sm = 0; sm < 2; ++sm)
#pragma unroll
        for (int sn = 0; sn < 4; ++sn)
#pragma unroll
            for (int r = 0; r < 8; ++r) {
                int row = m0 + wid * 32 + sm * 16 + r + lh * 8;
                int col = n0 + sn * 16 + ln;
                float val = acc[sm][sn][r];
                if (MODE == 0) {
                    int p = col / CC, rem = col % CC;
                    int hh = rem >> 6, dd = rem & 63;
                    int nn = row & (NN - 1);
                    if (p == 2) {
                        // V stored transposed: [B][H][D][N]
                        v[(((size_t)bb * HH + hh) * DD + dd) * NN + nn] = (bf16_t)val;
                    } else {
                        float sv = (p == 0) ? val * 8.0f : val;  // fold sqrt(D) into Q
                        bf16_t* dst = (p == 0) ? q : k;
                        dst[(((size_t)bb * HH + hh) * NN + nn) * DD + dd] = (bf16_t)sv;
                    }
                } else {
                    outf[(size_t)row * CC + col] = val;
                }
            }
}

// ---------------- flash attention: one block per (b, h, 64-query tile) ----------------
// Q,K: [B][H][N][D] bf16 (Q pre-scaled by 8).  V: [B][H][D][N] bf16 (transposed).
__global__ __launch_bounds__(256) void flash_kernel(const bf16_t* __restrict__ Q,
                                                    const bf16_t* __restrict__ K,
                                                    const bf16_t* __restrict__ Vtr,
                                                    bf16_t* __restrict__ O) {
    __shared__ __align__(16) bf16_t Qs[64 * 64];    // [q][d]
    __shared__ __align__(16) bf16_t Ks[64 * 64];    // [key][d]
    __shared__ __align__(16) bf16_t Vt[64 * 64];    // [d][key]
    __shared__ __align__(16) float  Ss[64 * 64];
    __shared__ __align__(16) bf16_t Ps[64 * 64];    // [q][key]
    __shared__ float red_s[4 * 64];                 // partial max / sum
    __shared__ float m_s[64], l_s[64], a_s[64];

    const int tid  = threadIdx.x;
    const int lane = tid & 31;
    const int wid  = tid >> 5;
    const int ln   = lane & 15;
    const int lh   = lane >> 4;
    const int wm   = wid & 3;   // 16-row group (queries)
    const int wn   = wid >> 2;  // 32-col group (keys for S, d for O)
    const int srow = tid & 63;  // softmax row
    const int sq   = tid >> 6;  // softmax quarter (16 cols)
    const int b    = blockIdx.z;
    const int h    = blockIdx.y;
    const int q0   = blockIdx.x * 64;

    const bf16_t* Qg = Q   + (((size_t)b * HH + h) * NN + q0) * DD;
    const bf16_t* Kg = K   + (((size_t)b * HH + h) * NN) * DD;
    const bf16_t* Vg = Vtr + (((size_t)b * HH + h) * DD) * NN;   // [d][n]

    // Q tile: contiguous b128 copy
#pragma unroll
    for (int i = 0; i < 2; ++i)
        ((uint4*)Qs)[tid + i * 256] = ((const uint4*)Qg)[tid + i * 256];
    if (tid < 64) { m_s[tid] = -1e30f; l_s[tid] = 0.f; }

    v8f o0 = (v8f){0.f, 0.f, 0.f, 0.f, 0.f, 0.f, 0.f, 0.f};
    v8f o1 = o0;

    const uint4* vg4 = (const uint4*)Vg;   // row stride NN/8 uint4

    for (int kt = 0; kt < NN / 64; ++kt) {
        const uint4* kg4 = (const uint4*)(Kg + (size_t)kt * 64 * DD);
        // K tile: contiguous b128 copy (tile is contiguous in global)
#pragma unroll
        for (int i = 0; i < 2; ++i)
            ((uint4*)Ks)[tid + i * 256] = kg4[tid + i * 256];
        // V tile: rows d=0..63, keys kt*64..+63 — b128 copies, row stride NN
#pragma unroll
        for (int i = 0; i < 2; ++i) {
            int t = tid + i * 256;
            int r = t >> 3, cu = t & 7;
            ((uint4*)Vt)[t] = vg4[(size_t)r * (NN / 8) + kt * 8 + cu];
        }
        // prefetch next K/V tiles behind this tile's compute
        if (kt + 1 < NN / 64) {
            __builtin_prefetch((const char*)(kg4 + 512) + tid * 32, 0, 0);
            __builtin_prefetch(&vg4[(size_t)(tid >> 3) * (NN / 8) + (kt + 1) * 8 + (tid & 7)], 0, 0);
        }
        __syncthreads();

        // S = Qs @ Ks^T : wave computes 16 rows x 32 key-cols; K-dim = D = 64
        v8f s0 = (v8f){0.f, 0.f, 0.f, 0.f, 0.f, 0.f, 0.f, 0.f};
        v8f s1 = s0;
#pragma unroll
        for (int ks = 0; ks < 2; ++ks) {
            int kb = ks * 32;
            v16bf af  = load_frag(Qs + (wm * 16 + ln) * 64 + kb, lh);
            v16bf bf0 = load_frag(Ks + (wn * 32 + ln) * 64 + kb, lh);
            v16bf bf1 = load_frag(Ks + (wn * 32 + 16 + ln) * 64 + kb, lh);
            s0 = wmma_bf16(af, bf0, s0);
            s1 = wmma_bf16(af, bf1, s1);
        }
#pragma unroll
        for (int r = 0; r < 8; ++r) {
            int row = wm * 16 + r + lh * 8;
            Ss[row * 64 + wn * 32 + ln]      = s0[r];
            Ss[row * 64 + wn * 32 + 16 + ln] = s1[r];
        }
        __syncthreads();

        // ---- online softmax, parallel over all 256 threads ----
        {   // phase A: partial max over 16 cols
            float pm = -1e30f;
#pragma unroll
            for (int j = 0; j < 16; ++j)
                pm = fmaxf(pm, Ss[srow * 64 + sq * 16 + j]);
            red_s[sq * 64 + srow] = pm;
        }
        __syncthreads();
        if (tid < 64) {   // phase B: combine max, compute alpha
            float mo = m_s[tid];
            float tm = fmaxf(fmaxf(red_s[tid], red_s[64 + tid]),
                             fmaxf(red_s[128 + tid], red_s[192 + tid]));
            tm = fmaxf(tm, mo);
            a_s[tid] = __expf(mo - tm);
            m_s[tid] = tm;
        }
        __syncthreads();
        {   // phase C: exponentiate, write packed P, partial sums
            float tm = m_s[srow];
            float sum = 0.f;
#pragma unroll
            for (int j = 0; j < 16; j += 2) {
                float p0 = __expf(Ss[srow * 64 + sq * 16 + j] - tm);
                float p1 = __expf(Ss[srow * 64 + sq * 16 + j + 1] - tm);
                sum += p0 + p1;
                ((u32*)Ps)[srow * 32 + sq * 8 + (j >> 1)] = pack_bf16(p0, p1);
            }
            red_s[sq * 64 + srow] = sum;
        }
        __syncthreads();
        if (tid < 64) {   // phase D: fold partial sums into running l
            l_s[tid] = a_s[tid] * l_s[tid] +
                       (red_s[tid] + red_s[64 + tid] + red_s[128 + tid] + red_s[192 + tid]);
        }

        // rescale O accumulators by alpha (a_s stable since phase B)
#pragma unroll
        for (int r = 0; r < 8; ++r) {
            float al = a_s[wm * 16 + r + lh * 8];
            o0[r] *= al;
            o1[r] *= al;
        }
        // O += P @ V : K-dim = 64 keys; V^T rows contiguous in Vt
#pragma unroll
        for (int ks = 0; ks < 2; ++ks) {
            int kb = ks * 32;
            v16bf af  = load_frag(Ps + (wm * 16 + ln) * 64 + kb, lh);
            v16bf bf0 = load_frag(Vt + (wn * 32 + ln) * 64 + kb, lh);
            v16bf bf1 = load_frag(Vt + (wn * 32 + 16 + ln) * 64 + kb, lh);
            o0 = wmma_bf16(af, bf0, o0);
            o1 = wmma_bf16(af, bf1, o1);
        }
        __syncthreads();
    }
    __syncthreads();   // l_s final values visible to all

    // epilogue: O[b][n][h*64 + d] = o / l
#pragma unroll
    for (int r = 0; r < 8; ++r) {
        int row  = wm * 16 + r + lh * 8;
        float inv = 1.0f / l_s[row];
        int grow = q0 + row;
        size_t base = ((size_t)b * NN + grow) * CC + h * DD;
        O[base + wn * 32 + ln]      = (bf16_t)(o0[r] * inv);
        O[base + wn * 32 + 16 + ln] = (bf16_t)(o1[r] * inv);
    }
}

// ---------------- host launcher ----------------
extern "C" void kernel_launch(void* const* d_in, const int* in_sizes, int n_in,
                              void* d_out, int out_size, void* d_ws, size_t ws_size,
                              hipStream_t stream) {
    (void)in_sizes; (void)n_in; (void)out_size; (void)ws_size;
    const float* x      = (const float*)d_in[0];  // [4,2048,768]
    const float* w_qkv  = (const float*)d_in[1];  // [768,2304]
    const float* w_proj = (const float*)d_in[2];  // [768,768]
    float* out = (float*)d_out;                   // [4,2048,768] f32

    char* ws = (char*)d_ws;
    size_t off = 0;
    auto alloc = [&](size_t bytes) -> void* {
        void* p = ws + off;
        off += bytes;
        off = (off + 255) & ~(size_t)255;
        return p;
    };

    bf16_t* xb      = (bf16_t*)alloc((size_t)MTOT * CC * 2);
    bf16_t* wqkvT   = (bf16_t*)alloc((size_t)QKVN * CC * 2);   // [2304][768]
    bf16_t* wprojT  = (bf16_t*)alloc((size_t)CC * CC * 2);     // [768][768]
    bf16_t* q       = (bf16_t*)alloc((size_t)BB * HH * NN * DD * 2);
    bf16_t* k       = (bf16_t*)alloc((size_t)BB * HH * NN * DD * 2);
    bf16_t* v       = (bf16_t*)alloc((size_t)BB * HH * DD * NN * 2);  // transposed
    bf16_t* ao      = (bf16_t*)alloc((size_t)MTOT * CC * 2);

    const int n_x = MTOT * CC;   // 6291456

    cvt_bf16_kernel<<<n_x / 4 / 256, 256, 0, stream>>>(x, xb, n_x);
    // transposing conversions: [768][N] f32 -> [N][768] bf16
    cvt_t_kernel<<<dim3(QKVN / 256, CC / 4), 256, 0, stream>>>(w_qkv,  wqkvT,  QKVN);
    cvt_t_kernel<<<dim3(CC   / 256, CC / 4), 256, 0, stream>>>(w_proj, wprojT, CC);

    // qkv = x @ w_qkv -> q,k [B,H,N,D] (q scaled by 8), v [B,H,D,N]
    gemm_bf16_kernel<QKVN, 0><<<dim3(QKVN / 64, MTOT / 256), 256, 0, stream>>>(
        xb, wqkvT, q, k, v, nullptr);

    // flash attention -> ao [B,N,C] bf16
    flash_kernel<<<dim3(NN / 64, HH, BB), 256, 0, stream>>>(q, k, v, ao);

    // out = ao @ w_proj (f32 output)
    gemm_bf16_kernel<CC, 1><<<dim3(CC / 64, MTOT / 256), 256, 0, stream>>>(
        ao, wprojT, nullptr, nullptr, nullptr, out);
}
